// GINLayer_45346264711281
// MI455X (gfx1250) — matrix operand order repaired and checked by reading, not compiled, not run on hardware
//
#include <hip/hip_runtime.h>
#include <hip/hip_bf16.h>

#define N_NODES 100000
#define N_EDGES 1600000
#define DIM 128
#define NUM_CLASSES 40

typedef __attribute__((ext_vector_type(16))) __bf16 v16bf;
typedef __attribute__((ext_vector_type(8)))  float  v8f;

union BfFrag {
    v16bf v;
    uint4 q[2];
};

// ---------------------------------------------------------------------------
// Zero-fill
// ---------------------------------------------------------------------------
__global__ void gin_fill_zero(float* __restrict__ p, long long n) {
    long long i = (long long)blockIdx.x * blockDim.x + threadIdx.x;
    if (i < n) p[i] = 0.0f;
}

// ---------------------------------------------------------------------------
// Edge scatter: agg[dst] += x[src].  One wave per edge; each lane moves a
// float4 chunk via 4 hardware global_atomic_add_f32 (resolved in L2).
// ---------------------------------------------------------------------------
static __device__ __forceinline__ void gatomic_fadd(float* p, float v) {
    asm volatile("global_atomic_add_f32 %0, %1, off"
                 : : "v"(p), "v"(v) : "memory");
}

__global__ __launch_bounds__(256) void gin_scatter(
    const float* __restrict__ xin, const int* __restrict__ edges,
    float* __restrict__ agg, int n_edges)
{
    long long gid = (long long)blockIdx.x * blockDim.x + threadIdx.x;
    int chunk = (int)(gid & 31);          // 32 chunks of 4 dims = 128
    long long e = gid >> 5;
    if (e >= n_edges) return;
    int src = edges[e * 2 + 0];
    int dst = edges[e * 2 + 1];
    const float4 v = *(const float4*)(xin + (size_t)src * DIM + chunk * 4);
    float* p = agg + (size_t)dst * DIM + chunk * 4;
    gatomic_fadd(p + 0, v.x);
    gatomic_fadd(p + 1, v.y);
    gatomic_fadd(p + 2, v.z);
    gatomic_fadd(p + 3, v.w);
}

// ---------------------------------------------------------------------------
// WMMA fragment-layout helpers (wave32, 16-bit operands).
//
// B layout (32x16 KxN tile, 8 VGPRs = 16 halves/lane):
//   lane = (n & 15) + 16*((k >> 3) & 1),  j = ((k & 31) >> 4)*8 + (k & 7)
// A layout (16x32 MxK tile):
//   lane = (m & 15) + 16*khalf, element j covers
//   k_local = ((j & 8) << 1) + khalf*8 + (j & 7)
// Weights / staged h are pre-swizzled in LDS so each lane's 16 halves are
// contiguous -> fragment load = 2x ds_load_b128.
// ---------------------------------------------------------------------------
static __device__ __forceinline__ int b_swizzle_idx(int k, int n) {
    int frag = (n >> 4) * 4 + (k >> 5);               // nt*4 + kc
    int lane = (n & 15) + (((k >> 3) & 1) << 4);
    int j    = (((k & 31) >> 4) << 3) | (k & 7);
    return (frag << 9) + (lane << 4) + j;
}
static __device__ __forceinline__ int a_swizzle_idx(int m, int k) {
    int kc = k >> 5, kl = k & 31;
    int lane = (m & 15) + (((kl >> 3) & 1) << 4);
    int j    = ((kl >> 4) << 3) | (kl & 7);
    return (kc << 9) + (lane << 4) + j;
}

// ---------------------------------------------------------------------------
// Layer-1 fused MLP:  x1 = relu( relu((agg+x) @ W1a + b1a) @ W1b + b1b )
// bf16 WMMA 16x16x32, f32 accumulate. 8 waves/block, 16 node-rows per wave.
// ---------------------------------------------------------------------------
__global__ __launch_bounds__(256) void gin_layer1_mlp(
    const float* __restrict__ x, const float* __restrict__ agg,
    const float* __restrict__ W1a, const float* __restrict__ b1a,
    const float* __restrict__ W1b, const float* __restrict__ b1b,
    float* __restrict__ x1, int n_nodes)
{
    __shared__ __align__(16) __bf16 sWa[DIM * DIM];   // 32 KB, B-swizzled
    __shared__ __align__(16) __bf16 sWb[DIM * DIM];   // 32 KB, B-swizzled
    __shared__ __align__(16) __bf16 sH[8][16 * DIM];  // 32 KB, A-swizzled
    __shared__ float sBa[DIM];
    __shared__ float sBb[DIM];

    const int t = threadIdx.x;
    for (int i = t; i < DIM * DIM; i += 256) {
        const int k = i >> 7, n = i & 127;
        const int w = b_swizzle_idx(k, n);
        sWa[w] = (__bf16)W1a[i];          // native f32->bf16 conversion
        sWb[w] = (__bf16)W1b[i];
    }
    for (int i = t; i < DIM; i += 256) { sBa[i] = b1a[i]; sBb[i] = b1b[i]; }
    __syncthreads();

    const int wave  = t >> 5;      // wave32
    const int lane  = t & 31;
    const int mrow  = lane & 15;
    const int khalf = lane >> 4;
    const int m0 = (blockIdx.x * 8 + wave) * 16;
    if (m0 >= n_nodes) return;     // wave-uniform: EXEC all-ones inside

    // ---- Build A fragments from agg + x (float4-vectorized) ----
    v16bf afrag[4];
    {
        const float* ra = agg + (size_t)(m0 + mrow) * DIM;
        const float* rx = x   + (size_t)(m0 + mrow) * DIM;
        for (int kc = 0; kc < 4; ++kc) {
            const int base = kc * 32 + khalf * 8;   // 32B aligned
            float va[16], vx[16];
            *(float4*)&va[0]  = *(const float4*)(ra + base);
            *(float4*)&va[4]  = *(const float4*)(ra + base + 4);
            *(float4*)&va[8]  = *(const float4*)(ra + base + 16);
            *(float4*)&va[12] = *(const float4*)(ra + base + 20);
            *(float4*)&vx[0]  = *(const float4*)(rx + base);
            *(float4*)&vx[4]  = *(const float4*)(rx + base + 4);
            *(float4*)&vx[8]  = *(const float4*)(rx + base + 16);
            *(float4*)&vx[12] = *(const float4*)(rx + base + 20);
            v16bf a;
#pragma unroll
            for (int j = 0; j < 16; ++j) a[j] = (__bf16)(va[j] + vx[j]);
            afrag[kc] = a;
        }
    }

    // ---- GEMM 1: h = relu(A @ W1a + b1a), stage h (bf16, A-layout) ----
    for (int nt = 0; nt < 8; ++nt) {
        v8f c = {};
        const int ncol = nt * 16 + mrow;
#pragma unroll
        for (int kc = 0; kc < 4; ++kc) {
            BfFrag B;
            const uint4* bp = (const uint4*)(sWa + (((nt << 2) + kc) << 9) + (lane << 4));
            B.q[0] = bp[0];
            B.q[1] = bp[1];
            c = __builtin_amdgcn_wmma_f32_16x16x32_bf16(
                    false, afrag[kc], false, B.v, (short)0, c, false, false);
        }
        const float bias = sBa[ncol];
#pragma unroll
        for (int i = 0; i < 8; ++i) {
            float v = c[i] + bias;
            v = v > 0.0f ? v : 0.0f;
            const int row = khalf * 8 + i;          // C layout: M index
            sH[wave][a_swizzle_idx(row, ncol)] = (__bf16)v;
        }
    }
    // wave-local LDS traffic is in-order; no cross-wave sharing -> no barrier.

    // ---- GEMM 2: x1 = relu(h @ W1b + b1b) ----
    v16bf hfrag[4];
#pragma unroll
    for (int kc = 0; kc < 4; ++kc) {
        BfFrag A;
        const uint4* hp = (const uint4*)(&sH[wave][0] + (kc << 9) + (lane << 4));
        A.q[0] = hp[0];
        A.q[1] = hp[1];
        hfrag[kc] = A.v;
    }
    for (int nt = 0; nt < 8; ++nt) {
        v8f c = {};
        const int ncol = nt * 16 + mrow;
#pragma unroll
        for (int kc = 0; kc < 4; ++kc) {
            BfFrag B;
            const uint4* bp = (const uint4*)(sWb + (((nt << 2) + kc) << 9) + (lane << 4));
            B.q[0] = bp[0];
            B.q[1] = bp[1];
            c = __builtin_amdgcn_wmma_f32_16x16x32_bf16(
                    false, hfrag[kc], false, B.v, (short)0, c, false, false);
        }
        const float bias = sBb[ncol];
#pragma unroll
        for (int i = 0; i < 8; ++i) {
            float v = c[i] + bias;
            v = v > 0.0f ? v : 0.0f;
            const int row = khalf * 8 + i;
            x1[(size_t)(m0 + row) * DIM + ncol] = v;
        }
    }
}

// ---------------------------------------------------------------------------
// Layer-2: logits = relu((agg2+x1) @ W2a + b2a) @ W2b + b2b, then softmax.
// Tiny GEMM (N=40): plain f32, one thread per node; weights in LDS.
// ---------------------------------------------------------------------------
__global__ __launch_bounds__(128) void gin_layer2_mlp(
    const float* __restrict__ x1, const float* __restrict__ agg2,
    const float* __restrict__ W2a, const float* __restrict__ b2a,
    const float* __restrict__ W2b, const float* __restrict__ b2b,
    float* __restrict__ out, int n_nodes)
{
    __shared__ float sWa[DIM * NUM_CLASSES];            // 20 KB
    __shared__ float sWb[NUM_CLASSES * NUM_CLASSES];    // 6.4 KB
    __shared__ float sBa[NUM_CLASSES];
    __shared__ float sBb[NUM_CLASSES];

    const int t = threadIdx.x;
    for (int i = t; i < DIM * NUM_CLASSES; i += 128) sWa[i] = W2a[i];
    for (int i = t; i < NUM_CLASSES * NUM_CLASSES; i += 128) sWb[i] = W2b[i];
    for (int i = t; i < NUM_CLASSES; i += 128) { sBa[i] = b2a[i]; sBb[i] = b2b[i]; }
    __syncthreads();

    const int node = blockIdx.x * blockDim.x + t;
    if (node >= n_nodes) return;

    const float* ra = agg2 + (size_t)node * DIM;
    const float* rx = x1   + (size_t)node * DIM;
    __builtin_prefetch(ra, 0, 3);   // global_prefetch_b8
    __builtin_prefetch(rx, 0, 3);

    float acc[NUM_CLASSES];
#pragma unroll
    for (int n = 0; n < NUM_CLASSES; ++n) acc[n] = sBa[n];
    for (int k = 0; k < DIM; ++k) {
        const float v = ra[k] + rx[k];
#pragma unroll
        for (int n = 0; n < NUM_CLASSES; ++n) acc[n] += v * sWa[k * NUM_CLASSES + n];
    }
#pragma unroll
    for (int n = 0; n < NUM_CLASSES; ++n) acc[n] = acc[n] > 0.0f ? acc[n] : 0.0f;

    float o[NUM_CLASSES];
#pragma unroll
    for (int n = 0; n < NUM_CLASSES; ++n) o[n] = sBb[n];
    for (int k = 0; k < NUM_CLASSES; ++k) {
        const float hk = acc[k];
#pragma unroll
        for (int n = 0; n < NUM_CLASSES; ++n) o[n] += hk * sWb[k * NUM_CLASSES + n];
    }

    float m = o[0];
#pragma unroll
    for (int n = 1; n < NUM_CLASSES; ++n) m = o[n] > m ? o[n] : m;
    float s = 0.0f;
#pragma unroll
    for (int n = 0; n < NUM_CLASSES; ++n) { o[n] = __expf(o[n] - m); s += o[n]; }
    const float inv = 1.0f / s;
    float* po = out + (size_t)node * NUM_CLASSES;
#pragma unroll
    for (int n = 0; n < NUM_CLASSES; ++n) po[n] = o[n] * inv;
}

// ---------------------------------------------------------------------------
extern "C" void kernel_launch(void* const* d_in, const int* in_sizes, int n_in,
                              void* d_out, int out_size, void* d_ws, size_t ws_size,
                              hipStream_t stream) {
    const float* x0  = (const float*)d_in[0];   // node_embeddings
    const int*   edg = (const int*)  d_in[1];   // edge_index (E,2)
    const float* W1a = (const float*)d_in[2];
    const float* b1a = (const float*)d_in[3];
    const float* W1b = (const float*)d_in[4];
    const float* b1b = (const float*)d_in[5];
    const float* W2a = (const float*)d_in[6];
    const float* b2a = (const float*)d_in[7];
    const float* W2b = (const float*)d_in[8];
    const float* b2b = (const float*)d_in[9];
    float* out = (float*)d_out;

    const size_t per = (size_t)N_NODES * DIM;   // elements per feature buffer
    float* agg1 = (float*)d_ws;
    float* x1   = agg1 + per;
    float* agg2 = x1 + per;

    const long long fillN = (long long)per;
    const int fillBlocks = (int)((fillN + 255) / 256);
    gin_fill_zero<<<fillBlocks, 256, 0, stream>>>(agg1, fillN);
    gin_fill_zero<<<fillBlocks, 256, 0, stream>>>(agg2, fillN);

    const long long sthreads = (long long)N_EDGES * 32;
    const int sblocks = (int)((sthreads + 255) / 256);

    // Layer 1
    gin_scatter<<<sblocks, 256, 0, stream>>>(x0, edg, agg1, N_EDGES);
    const int l1blocks = (N_NODES / 16 + 7) / 8;   // 8 wave-tiles of 16 rows
    gin_layer1_mlp<<<l1blocks, 256, 0, stream>>>(x0, agg1, W1a, b1a, W1b, b1b,
                                                 x1, N_NODES);
    // Layer 2
    gin_scatter<<<sblocks, 256, 0, stream>>>(x1, edg, agg2, N_EDGES);
    gin_layer2_mlp<<<(N_NODES + 127) / 128, 128, 0, stream>>>(
        x1, agg2, W2a, b2a, W2b, b2b, out, N_NODES);
}
